// TransformerBlock_4922032521490
// MI455X (gfx1250) — matrix operand order repaired and checked by reading, not compiled, not run on hardware
//
#include <hip/hip_runtime.h>
#include <math.h>

// ---------------- problem constants ----------------
#define DIMC   1024
#define SEQ    2048
#define BATCH  2
#define NROWS  (BATCH*SEQ)       // 4096
#define DFFC   4096
#define HEADS  16
#define HDIM   64

// ---------------- WMMA types ----------------
typedef __attribute__((ext_vector_type(16))) __bf16 v16bf;
typedef __attribute__((ext_vector_type(8)))  float  v8f;
typedef __attribute__((ext_vector_type(4)))  int    v4i;

union AF { v16bf v; unsigned u[8]; };   // bf16 fragment as 8 packed dwords

typedef unsigned short bhalf;           // bf16 bit pattern

__device__ __forceinline__ bhalf f2bf(float f) {
    unsigned u = __float_as_uint(f);
    unsigned r = 0x7FFFu + ((u >> 16) & 1u);   // round-to-nearest-even
    return (bhalf)((u + r) >> 16);
}

// ---------------- CDNA5 async global->LDS copy (guarded) ----------------
#if __has_builtin(__builtin_amdgcn_global_load_async_to_lds_b128) && \
    __has_builtin(__builtin_amdgcn_s_wait_asynccnt)
#define HAVE_ASYNC_LDS 1
typedef __attribute__((address_space(1))) v4i* g_v4p;
typedef __attribute__((address_space(3))) v4i* l_v4p;
#endif

__device__ __forceinline__ void copy16_g2l(bhalf* lds, const bhalf* g) {
#ifdef HAVE_ASYNC_LDS
    __builtin_amdgcn_global_load_async_to_lds_b128((g_v4p)g, (l_v4p)lds, 0, 0);
#else
    *(uint4*)lds = *(const uint4*)g;
#endif
}
__device__ __forceinline__ void wait_async_copies() {
#ifdef HAVE_ASYNC_LDS
    __builtin_amdgcn_s_wait_asynccnt(0);
#endif
}

// =====================================================================
// fp32 -> bf16 conversion kernels (one-shot, bandwidth bound)
// =====================================================================
__global__ __launch_bounds__(256)
void cvt_bf16_kernel(const float* __restrict__ src, bhalf* __restrict__ dst, long n)
{
    long i = ((long)blockIdx.x * 256 + threadIdx.x) * 8;
    if (i + 8 > n) return;
    float4 f0 = *(const float4*)(src + i);
    float4 f1 = *(const float4*)(src + i + 4);
    uint4 o;
    o.x = (unsigned)f2bf(f0.x) | ((unsigned)f2bf(f0.y) << 16);
    o.y = (unsigned)f2bf(f0.z) | ((unsigned)f2bf(f0.w) << 16);
    o.z = (unsigned)f2bf(f1.x) | ((unsigned)f2bf(f1.y) << 16);
    o.w = (unsigned)f2bf(f1.z) | ((unsigned)f2bf(f1.w) << 16);
    *(uint4*)(dst + i) = o;
}

// transpose + convert: src fp32 R x C  ->  dst bf16 C x R   (batched via z)
__global__ __launch_bounds__(256)
void cvt_t_bf16_kernel(const float* __restrict__ src, bhalf* __restrict__ dst,
                       int R, int C, long sSrc, long sDst)
{
    __shared__ bhalf tile[64 * 66];
    src += (long)blockIdx.z * sSrc;
    dst += (long)blockIdx.z * sDst;
    const int r0 = blockIdx.y * 64, c0 = blockIdx.x * 64;
    const int tid = threadIdx.x;
    #pragma unroll
    for (int it = 0; it < 16; ++it) {
        int idx = tid + it * 256;
        int r = idx >> 6, c = idx & 63;
        tile[r * 66 + c] = f2bf(src[(long)(r0 + r) * C + c0 + c]);
    }
    __syncthreads();
    #pragma unroll
    for (int it = 0; it < 16; ++it) {
        int idx = tid + it * 256;
        int r = idx & 63, c = idx >> 6;
        dst[(long)(c0 + c) * R + r0 + r] = tile[r * 66 + c];
    }
}

// =====================================================================
// Tiled bf16 WMMA GEMM: C[z] = act(A[z] @ B[z]^T + bias) (+ residual[z])
// A: M x K bf16 row-major.  B: N x K bf16 row-major (pre-transposed).
// BM=BN=128, BK=64. 256 threads = 8 waves; wave tile 32x64 (2x4 WMMA tiles),
// 16 v_wmma per K-step. Staging via async global->LDS b128 when available.
// =====================================================================
#define BM 128
#define BN 128
#define BK 64
#define LDT 72   // BK + 8 pad (bhalf units); row stride 144B (16B aligned)

__global__ __launch_bounds__(256)
void gemm_bf16_kernel(const bhalf* __restrict__ A, const bhalf* __restrict__ B,
                      float* __restrict__ C,
                      const float* __restrict__ bias,
                      const float* __restrict__ residual,
                      int M, int N, int K,
                      long sA, long sB, long sC, long sR, int act)
{
    __shared__ bhalf As[BM * LDT];
    __shared__ bhalf Bs[BN * LDT];   // [n][k]

    const int z = blockIdx.z;
    A += (long)z * sA;  B += (long)z * sB;  C += (long)z * sC;
    const float* R = residual ? residual + (long)z * sR : nullptr;

    const int bm   = blockIdx.y * BM;
    const int bn   = blockIdx.x * BN;
    const int tid  = threadIdx.x;
    const int wave = tid >> 5;
    const int lane = tid & 31;
    const int half = lane >> 4;
    const int l16  = lane & 15;
    const int wm   = (wave & 3) * 32;   // 4 waves along M
    const int wn   = (wave >> 2) * 64;  // 2 waves along N

    const v8f vzero = {0.f,0.f,0.f,0.f,0.f,0.f,0.f,0.f};
    v8f acc[2][4];
    #pragma unroll
    for (int tm = 0; tm < 2; ++tm)
        #pragma unroll
        for (int tn = 0; tn < 4; ++tn) acc[tm][tn] = vzero;

    for (int k0 = 0; k0 < K; k0 += BK) {
        // ---- stage A and B tiles (128 x 64 bf16 each), 16B per transfer ----
        #pragma unroll
        for (int it = 0; it < 4; ++it) {
            int idx = tid + it * 256;        // 1024 transfers per tile
            int row = idx >> 3;
            int q   = (idx & 7) * 8;
            copy16_g2l(&As[row * LDT + q], A + (long)(bm + row) * K + k0 + q);
            copy16_g2l(&Bs[row * LDT + q], B + (long)(bn + row) * K + k0 + q);
        }
        wait_async_copies();
        __syncthreads();

        // ---- 2 WMMA K-steps (ISA bf16 A 16x32 / B 32x16 fragment layouts) ----
        #pragma unroll
        for (int ks = 0; ks < 2; ++ks) {
            AF a0, a1;
            {
                const bhalf* b0 = &As[(wm + l16) * LDT];
                const bhalf* b1 = &As[(wm + 16 + l16) * LDT];
                #pragma unroll
                for (int j = 0; j < 8; ++j) {
                    int k = ks * 32 + ((j < 4) ? 2 * j : 16 + 2 * (j - 4)) + 8 * half;
                    a0.u[j] = *(const unsigned*)(b0 + k);
                    a1.u[j] = *(const unsigned*)(b1 + k);
                }
            }
            #pragma unroll
            for (int tn = 0; tn < 4; ++tn) {
                AF bb;
                const bhalf* base = &Bs[(wn + tn * 16 + l16) * LDT] + ks * 32;
                #pragma unroll
                for (int j = 0; j < 8; ++j)
                    bb.u[j] = *(const unsigned*)(base + ((j < 4) ? 2 * j : 16 + 2 * (j - 4)) + 8 * half);
                acc[0][tn] = __builtin_amdgcn_wmma_f32_16x16x32_bf16(
                    false, a0.v, false, bb.v, (short)0, acc[0][tn], false, false);
                acc[1][tn] = __builtin_amdgcn_wmma_f32_16x16x32_bf16(
                    false, a1.v, false, bb.v, (short)0, acc[1][tn], false, false);
            }
        }
        __syncthreads();
    }

    // ---- epilogue: bias / GELU / residual ----
    #pragma unroll
    for (int tm = 0; tm < 2; ++tm)
        #pragma unroll
        for (int tn = 0; tn < 4; ++tn) {
            int n = bn + wn + tn * 16 + l16;
            float bv = bias ? bias[n] : 0.0f;
            #pragma unroll
            for (int i = 0; i < 8; ++i) {
                int m = bm + wm + tm * 16 + i + half * 8;
                float v = acc[tm][tn][i] + bv;
                if (act == 1) v = 0.5f * v * (1.0f + erff(v * 0.70710678118654752f));
                if (R) v += R[(long)m * N + n];
                C[(long)m * N + n] = v;
            }
        }
}

// =====================================================================
// LayerNorm over rows of length C. One block per row.
// =====================================================================
__global__ __launch_bounds__(256)
void layernorm_kernel(const float* __restrict__ x, const float* __restrict__ g,
                      const float* __restrict__ b, float* __restrict__ y, int C)
{
    __shared__ float red[256];
    const int tid = threadIdx.x;
    const float* xr = x + (long)blockIdx.x * C;
    float s = 0.f, ss = 0.f;
    for (int c = tid; c < C; c += 256) { float v = xr[c]; s += v; ss += v * v; }
    red[tid] = s; __syncthreads();
    for (int st = 128; st > 0; st >>= 1) { if (tid < st) red[tid] += red[tid + st]; __syncthreads(); }
    float mean = red[0] / C; __syncthreads();
    red[tid] = ss; __syncthreads();
    for (int st = 128; st > 0; st >>= 1) { if (tid < st) red[tid] += red[tid + st]; __syncthreads(); }
    float var = red[0] / C - mean * mean;
    float inv = rsqrtf(var + 1e-5f);
    float* yr = y + (long)blockIdx.x * C;
    for (int c = tid; c < C; c += 256) yr[c] = (xr[c] - mean) * inv * g[c] + b[c];
}

// =====================================================================
// theta + softmax over rows of length SEQ, in place. One block per row.
// theta = 0.5 + 0.2*sigmoid(s) + 0.15*tanh(s) + 0.1*relu(s)
// =====================================================================
__global__ __launch_bounds__(256)
void theta_softmax_kernel(float* __restrict__ S)
{
    __shared__ float buf[SEQ];
    __shared__ float red[256];
    const int tid = threadIdx.x;
    float* row = S + (long)blockIdx.x * SEQ;
    float mx = -1e30f;
    for (int c = tid; c < SEQ; c += 256) {
        float q   = row[c];
        float sig = 1.0f / (1.0f + __expf(-q));
        float th  = 0.5f + 0.2f * sig + 0.15f * tanhf(q) + 0.1f * fmaxf(q, 0.0f);
        buf[c] = th;
        mx = fmaxf(mx, th);
    }
    red[tid] = mx; __syncthreads();
    for (int st = 128; st > 0; st >>= 1) { if (tid < st) red[tid] = fmaxf(red[tid], red[tid + st]); __syncthreads(); }
    mx = red[0]; __syncthreads();
    float sum = 0.f;
    for (int c = tid; c < SEQ; c += 256) { float e = __expf(buf[c] - mx); buf[c] = e; sum += e; }
    red[tid] = sum; __syncthreads();
    for (int st = 128; st > 0; st >>= 1) { if (tid < st) red[tid] += red[tid + st]; __syncthreads(); }
    float inv = 1.0f / red[0];
    for (int c = tid; c < SEQ; c += 256) row[c] = buf[c] * inv;
}

// =====================================================================
// Flash-style MHA: one WG (4 waves / 128 thr) per (64-row block, head, batch).
// qkv layout per row: [q(1024) | k(1024) | v(1024)], head h at h*64.
// =====================================================================
#define FB 64
#define LQ 72    // 64 + 8 pad (bhalf)
#define LSS 66

__global__ __launch_bounds__(128)
void mha_flash_kernel(const float* __restrict__ qkv, float* __restrict__ out)
{
    __shared__ bhalf Qs[FB * LQ];
    __shared__ bhalf Ks[FB * LQ];     // Ks[m][d]
    __shared__ bhalf Ps[FB * LQ];     // Ps[row][key]
    __shared__ bhalf Vt[FB * LQ];     // Vt[d][m]
    __shared__ float Ss[FB * LSS];
    __shared__ float mrow[FB], lrow[FB], arow[FB];

    const int b = blockIdx.z, h = blockIdx.y, rb = blockIdx.x;
    const int tid = threadIdx.x, wave = tid >> 5, lane = tid & 31;
    const int half = lane >> 4, l16 = lane & 15;
    const long rowbase = (long)b * SEQ + (long)rb * FB;
    const long qoff = rowbase * (3 * DIMC) + h * HDIM;

    for (int i = tid; i < FB * HDIM; i += 128) {
        int r = i >> 6, d = i & 63;
        Qs[r * LQ + d] = f2bf(qkv[qoff + (long)r * (3 * DIMC) + d] * 0.125f);  // *HDIM^-0.5
    }
    if (tid < FB) { mrow[tid] = -1e30f; lrow[tid] = 0.f; }
    __syncthreads();

    AF qf[2];
    {
        const bhalf* base = &Qs[(wave * 16 + l16) * LQ];
        #pragma unroll
        for (int ks = 0; ks < 2; ++ks)
            #pragma unroll
            for (int j = 0; j < 8; ++j) {
                int k = ks * 32 + ((j < 4) ? 2 * j : 16 + 2 * (j - 4)) + 8 * half;
                qf[ks].u[j] = *(const unsigned*)(base + k);
            }
    }

    const v8f vzero = {0.f,0.f,0.f,0.f,0.f,0.f,0.f,0.f};
    v8f oacc[4];
    #pragma unroll
    for (int td = 0; td < 4; ++td) oacc[td] = vzero;

    for (int j0 = 0; j0 < SEQ; j0 += FB) {
        for (int i = tid; i < FB * HDIM; i += 128) {
            int m = i >> 6, d = i & 63;
            long base = ((long)b * SEQ + j0 + m) * (3 * DIMC) + h * HDIM + d;
            Ks[m * LQ + d] = f2bf(qkv[base + DIMC]);       // k
            Vt[d * LQ + m] = f2bf(qkv[base + 2 * DIMC]);   // v transposed
        }
        __syncthreads();

        v8f sacc[4];
        #pragma unroll
        for (int tn = 0; tn < 4; ++tn) sacc[tn] = vzero;
        #pragma unroll
        for (int tn = 0; tn < 4; ++tn) {
            const bhalf* nb = &Ks[(tn * 16 + l16) * LQ];
            #pragma unroll
            for (int ks = 0; ks < 2; ++ks) {
                AF bf;
                const bhalf* base = nb + ks * 32;
                #pragma unroll
                for (int j = 0; j < 8; ++j) bf.u[j] = *(const unsigned*)(base + ((j < 4) ? 2 * j : 16 + 2 * (j - 4)) + 8 * half);
                sacc[tn] = __builtin_amdgcn_wmma_f32_16x16x32_bf16(
                    false, qf[ks].v, false, bf.v, (short)0, sacc[tn], false, false);
            }
        }
        #pragma unroll
        for (int tn = 0; tn < 4; ++tn)
            #pragma unroll
            for (int i = 0; i < 8; ++i)
                Ss[(wave * 16 + i + half * 8) * LSS + tn * 16 + l16] = sacc[tn][i];
        __syncthreads();

        if (tid < FB) {
            int r = tid;
            float om = mrow[r], cm = om;
            for (int c = 0; c < FB; ++c) cm = fmaxf(cm, Ss[r * LSS + c]);
            float al = __expf(om - cm);
            float sum = 0.f;
            for (int c = 0; c < FB; ++c) {
                float p = __expf(Ss[r * LSS + c] - cm);
                Ps[r * LQ + c] = f2bf(p);
                sum += p;
            }
            mrow[r] = cm; lrow[r] = al * lrow[r] + sum; arow[r] = al;
        }
        __syncthreads();

        float av[8];
        #pragma unroll
        for (int i = 0; i < 8; ++i) av[i] = arow[wave * 16 + i + half * 8];
        AF pf[2];
        {
            const bhalf* base = &Ps[(wave * 16 + l16) * LQ];
            #pragma unroll
            for (int ks = 0; ks < 2; ++ks)
                #pragma unroll
                for (int j = 0; j < 8; ++j) {
                    int k = ks * 32 + ((j < 4) ? 2 * j : 16 + 2 * (j - 4)) + 8 * half;
                    pf[ks].u[j] = *(const unsigned*)(base + k);
                }
        }
        #pragma unroll
        for (int td = 0; td < 4; ++td) {
            #pragma unroll
            for (int i = 0; i < 8; ++i) oacc[td][i] *= av[i];
            const bhalf* nb = &Vt[(td * 16 + l16) * LQ];
            #pragma unroll
            for (int ks = 0; ks < 2; ++ks) {
                AF bf;
                const bhalf* base = nb + ks * 32;
                #pragma unroll
                for (int j = 0; j < 8; ++j) bf.u[j] = *(const unsigned*)(base + ((j < 4) ? 2 * j : 16 + 2 * (j - 4)) + 8 * half);
                oacc[td] = __builtin_amdgcn_wmma_f32_16x16x32_bf16(
                    false, pf[ks].v, false, bf.v, (short)0, oacc[td], false, false);
            }
        }
        __syncthreads();
    }

    float linv[8];
    #pragma unroll
    for (int i = 0; i < 8; ++i) linv[i] = 1.0f / lrow[wave * 16 + i + half * 8];
    #pragma unroll
    for (int td = 0; td < 4; ++td)
        #pragma unroll
        for (int i = 0; i < 8; ++i) {
            int m = wave * 16 + i + half * 8;
            int d = td * 16 + l16;
            out[(rowbase + m) * DIMC + h * HDIM + d] = oacc[td][i] * linv[i];
        }
}

// =====================================================================
// Host orchestration
// =====================================================================
static void launch_gemm(const bhalf* A, const bhalf* B, float* C,
                        const float* bias, const float* residual,
                        int M, int N, int K,
                        long sA, long sB, long sC, long sR, int nz,
                        int act, hipStream_t s)
{
    dim3 grid(N / BN, M / BM, nz), blk(256);
    gemm_bf16_kernel<<<grid, blk, 0, s>>>(A, B, C, bias, residual, M, N, K,
                                          sA, sB, sC, sR, act);
}

static void launch_cvt(const float* src, bhalf* dst, long n, hipStream_t s)
{
    long blocks = n / (8 * 256);
    cvt_bf16_kernel<<<(unsigned)blocks, 256, 0, s>>>(src, dst, n);
}

static void launch_cvt_t(const float* src, bhalf* dst, int R, int C,
                         long sSrc, long sDst, int nz, hipStream_t s)
{
    dim3 grid(C / 64, R / 64, nz), blk(256);
    cvt_t_bf16_kernel<<<grid, blk, 0, s>>>(src, dst, R, C, sSrc, sDst);
}

extern "C" void kernel_launch(void* const* d_in, const int* in_sizes, int n_in,
                              void* d_out, int out_size, void* d_ws, size_t ws_size,
                              hipStream_t stream)
{
    const float* x      = (const float*)d_in[0];
    const float* ln1g   = (const float*)d_in[1];
    const float* ln1b   = (const float*)d_in[2];
    const float* qkv_w  = (const float*)d_in[3];
    const float* qkv_b  = (const float*)d_in[4];
    const float* proj_w = (const float*)d_in[5];
    const float* proj_b = (const float*)d_in[6];
    const float* kq_w   = (const float*)d_in[7];
    const float* kq_b   = (const float*)d_in[8];
    const float* kk_w   = (const float*)d_in[9];
    const float* kk_b   = (const float*)d_in[10];
    const float* kv_w   = (const float*)d_in[11];
    const float* kv_b   = (const float*)d_in[12];
    const float* ko_w   = (const float*)d_in[13];
    const float* ko_b   = (const float*)d_in[14];
    const float* ln2g   = (const float*)d_in[15];
    const float* ln2b   = (const float*)d_in[16];
    const float* fc1_w  = (const float*)d_in[17];
    const float* fc1_b  = (const float*)d_in[18];
    const float* fc2_w  = (const float*)d_in[19];
    const float* fc2_b  = (const float*)d_in[20];
    float* out = (float*)d_out;
    float* ws  = (float*)d_ws;

    const size_t MI = 1024 * 1024;
    const size_t M4 = (size_t)NROWS * DIMC;        // 4 Mi elements

    // fp32 region (40 Mi floats)
    float* bufA   = ws;                  // h0 / kao
    float* bufQ   = ws + 4  * MI;        // qkv (12Mi) -> kq,kk,kv
    float* bufATT = ws + 16 * MI;        // att / x2
    float* bufX1  = ws + 20 * MI;        // x1 / h
    float* bufSC  = ws + 24 * MI;        // scores (8Mi) / h1 (16Mi)
    float* kq = bufQ, *kk = bufQ + M4, *kv = bufQ + 2 * M4;

    // bf16 region (44 Mi bhalf) after fp32 region
    bhalf* bb = (bhalf*)(ws + 40 * MI);
    bhalf* wT_qkv = bb;                  // 3072 x 1024
    bhalf* wT_proj= bb + 3  * MI;        // 1024 x 1024
    bhalf* wT_kq  = bb + 4  * MI;
    bhalf* wT_kk  = bb + 5  * MI;
    bhalf* wT_kv  = bb + 6  * MI;
    bhalf* wT_ko  = bb + 7  * MI;
    bhalf* wT_fc1 = bb + 8  * MI;        // 4096 x 1024
    bhalf* wT_fc2 = bb + 12 * MI;        // 1024 x 4096
    bhalf* actA   = bb + 16 * MI;        // rotating A operand (<= 16 Mi)
    bhalf* actB   = bb + 32 * MI;        // 4 Mi
    bhalf* actC   = bb + 36 * MI;        // 4 Mi
    bhalf* actD   = bb + 40 * MI;        // 4 Mi (kv^T bf16)

    const long sRow = (long)SEQ * DIMC;
    const long sScr = (long)SEQ * SEQ;

    // ---- one-shot weight transpose-converts (B operands are N x K) ----
    launch_cvt_t(qkv_w, wT_qkv, DIMC, 3 * DIMC, 0, 0, 1, stream);
    launch_cvt_t(proj_w, wT_proj, DIMC, DIMC, 0, 0, 1, stream);
    launch_cvt_t(kq_w,  wT_kq,  DIMC, DIMC, 0, 0, 1, stream);
    launch_cvt_t(kk_w,  wT_kk,  DIMC, DIMC, 0, 0, 1, stream);
    launch_cvt_t(kv_w,  wT_kv,  DIMC, DIMC, 0, 0, 1, stream);
    launch_cvt_t(ko_w,  wT_ko,  DIMC, DIMC, 0, 0, 1, stream);
    launch_cvt_t(fc1_w, wT_fc1, DIMC, DFFC, 0, 0, 1, stream);
    launch_cvt_t(fc2_w, wT_fc2, DFFC, DIMC, 0, 0, 1, stream);

    // ---- x1 = x + proj(MHA(LN1(x))) ----
    layernorm_kernel<<<NROWS, 256, 0, stream>>>(x, ln1g, ln1b, bufA, DIMC);
    launch_cvt(bufA, actA, M4, stream);
    launch_gemm(actA, wT_qkv, bufQ, qkv_b, nullptr, NROWS, 3 * DIMC, DIMC,
                0, 0, 0, 0, 1, 0, stream);
    mha_flash_kernel<<<dim3(SEQ / FB, HEADS, BATCH), 128, 0, stream>>>(bufQ, bufATT);
    launch_cvt(bufATT, actA, M4, stream);
    launch_gemm(actA, wT_proj, bufX1, proj_b, x, NROWS, DIMC, DIMC,
                0, 0, 0, 0, 1, 0, stream);

    // ---- kernel attention: x2 = x1 + ko(softmax(theta(q k^T)) @ v) ----
    launch_cvt(bufX1, actA, M4, stream);
    launch_gemm(actA, wT_kq, kq, kq_b, nullptr, NROWS, DIMC, DIMC, 0, 0, 0, 0, 1, 0, stream);
    launch_gemm(actA, wT_kk, kk, kk_b, nullptr, NROWS, DIMC, DIMC, 0, 0, 0, 0, 1, 0, stream);
    launch_gemm(actA, wT_kv, kv, kv_b, nullptr, NROWS, DIMC, DIMC, 0, 0, 0, 0, 1, 0, stream);
    launch_cvt(kq, actB, M4, stream);                                // A of qk GEMM
    launch_cvt(kk, actC, M4, stream);                                // B (N x K natural)
    launch_cvt_t(kv, actD, SEQ, DIMC, sRow, (long)DIMC * SEQ, BATCH, stream); // v^T per batch
    launch_gemm(actB, actC, bufSC, nullptr, nullptr, SEQ, SEQ, DIMC,
                sRow, sRow, sScr, 0, BATCH, 0, stream);
    theta_softmax_kernel<<<BATCH * SEQ, 256, 0, stream>>>(bufSC);
    launch_cvt(bufSC, actA, (long)BATCH * SEQ * SEQ, stream);
    launch_gemm(actA, actD, bufA, nullptr, nullptr, SEQ, DIMC, SEQ,
                sScr, (long)DIMC * SEQ, sRow, 0, BATCH, 0, stream);
    launch_cvt(bufA, actB, M4, stream);
    launch_gemm(actB, wT_ko, bufATT, ko_b, bufX1, NROWS, DIMC, DIMC,
                0, 0, 0, 0, 1, 0, stream);                 // x2 in bufATT

    // ---- out = x2 + fc2(gelu(fc1(LN2(x2)))) ----
    layernorm_kernel<<<NROWS, 256, 0, stream>>>(bufATT, ln2g, ln2b, bufX1, DIMC);
    launch_cvt(bufX1, actA, M4, stream);
    launch_gemm(actA, wT_fc1, bufSC, fc1_b, nullptr, NROWS, DFFC, DIMC,
                0, 0, 0, 0, 1, /*gelu*/1, stream);
    launch_cvt(bufSC, actA, (long)NROWS * DFFC, stream);
    launch_gemm(actA, wT_fc2, out, fc2_b, bufATT, NROWS, DIMC, DFFC,
                0, 0, 0, 0, 1, 0, stream);
}
